// FFTNet_60155311947873
// MI455X (gfx1250) — matrix operand order repaired and checked by reading, not compiled, not run on hardware
//
#include <hip/hip_runtime.h>
#include <math.h>

typedef _Float16 half_t;
typedef _Float16 v16h __attribute__((ext_vector_type(16)));
typedef _Float16 v8h  __attribute__((ext_vector_type(8)));
typedef float    v8f  __attribute__((ext_vector_type(8)));

#define IMG   40
#define FREQ  (IMG * IMG)      // 1600
#define BATCH 256
#define LDSW  64               // padded K width (128B rows -> aligned b128)
#define TILE_ROWS 48           // padded 40 -> 3 x 16
#define IMGS_PER_BLK 4
#define WP_BIN   (2 * TILE_ROWS * LDSW)   // halves per bin in packed weights: 6144
#define WP_PLANE (TILE_ROWS * LDSW)       // 3072

// ---------------------------------------------------------------------------
// DFT matrix F (ortho): f[0..1600)=Re(F), [1600..3200)=Im(F), [3200..4800)=Im(F*)
// ---------------------------------------------------------------------------
__global__ void init_dft_kernel(half_t* __restrict__ f) {
    int idx = blockIdx.x * blockDim.x + threadIdx.x;
    if (idx >= FREQ) return;
    int u = idx / IMG, h = idx % IMG;
    int t = (u * h) % IMG;
    float ang = -6.283185307179586f * (float)t / (float)IMG;
    float sc  = 0.15811388300841897f;   // 1/sqrt(40)
    float sn, cs;
    __sincosf(ang, &sn, &cs);
    f[idx]            = (half_t)(cs * sc);
    f[FREQ + idx]     = (half_t)(sn * sc);
    f[2 * FREQ + idx] = (half_t)(-sn * sc);
}

__global__ void pack_kernel(const float* __restrict__ x,
                            half_t* __restrict__ re, half_t* __restrict__ im, int n) {
    int i = blockIdx.x * blockDim.x + threadIdx.x;
    if (i < n) { re[i] = (half_t)x[i]; im[i] = (half_t)0.f; }
}

// ---------------------------------------------------------------------------
// Repack layer weights W[i][o][f][2] (f32) -> Wp[f][p][o(48)][i(64)] (f16, 0-pad).
// One thread per (o, f) row: reads coalesced along f, writes aligned v8h rows.
// ---------------------------------------------------------------------------
__global__ void repack_w_kernel(const float* __restrict__ W, half_t* __restrict__ Wp,
                                int ci, int co) {
    int t = blockIdx.x * blockDim.x + threadIdx.x;
    if (t >= FREQ * TILE_ROWS) return;
    int f = t % FREQ;
    int o = t / FREQ;                       // 0..47
    int oc = o < co ? o : co - 1;           // clamped (value masked below)
#pragma unroll
    for (int p = 0; p < 2; ++p) {
        v8h* dstv = (v8h*)&Wp[f * WP_BIN + p * WP_PLANE + o * LDSW];
#pragma unroll
        for (int c = 0; c < 8; ++c) {
            v8h v;
#pragma unroll
            for (int e = 0; e < 8; ++e) {
                int i   = c * 8 + e;
                int icl = i < ci ? i : ci - 1;
                float val = W[((icl * co + oc) * FREQ + f) * 2 + p];  // always safe
                v[e] = (i < ci && o < co) ? (half_t)val : (half_t)0.f;
            }
            dstv[c] = v;
        }
    }
}

// ---- fragment helpers ------------------------------------------------------
// A-layout (16-bit A 16x32): lane row = lane&15, K runs [k0..k0+7],[k0+16..k0+23]
__device__ inline v16h ld_fragA(const half_t* rowBase, int k0) {
    v8h lo = *(const v8h*)(rowBase + k0);
    v8h hi = *(const v8h*)(rowBase + k0 + 16);
    v16h r;
#pragma unroll
    for (int e = 0; e < 8; ++e) { r[e] = lo[e]; r[e + 8] = hi[e]; }
    return r;
}
// B-layout (16-bit B 32x16): lane col = lane&15, K runs [k0..k0+7],[k0+16..k0+23]
__device__ inline v16h ld_fragB(const half_t* rowBase, int k0) {
    v8h lo = *(const v8h*)(rowBase + k0);
    v8h hi = *(const v8h*)(rowBase + k0 + 16);
    v16h r;
#pragma unroll
    for (int e = 0; e < 8; ++e) { r[e] = lo[e]; r[e + 8] = hi[e]; }
    return r;
}
__device__ inline v8h packv(v8f a) {
    v8h p;
#pragma unroll
    for (int e = 0; e < 8; ++e) p[e] = (half_t)a[e];
    return p;
}
#define CWMMA(AR, AI, AIN, BR, BI, CR, CI)                                              \
    CR = __builtin_amdgcn_wmma_f32_16x16x32_f16(false, AR,  false, BR, (short)0, CR, false, false); \
    CR = __builtin_amdgcn_wmma_f32_16x16x32_f16(false, AIN, false, BI, (short)0, CR, false, false); \
    CI = __builtin_amdgcn_wmma_f32_16x16x32_f16(false, AR,  false, BI, (short)0, CI, false, false); \
    CI = __builtin_amdgcn_wmma_f32_16x16x32_f16(false, AI,  false, BR, (short)0, CI, false, false);

// ---------------------------------------------------------------------------
// Fused per-image 2D (i)FFT: out = F . z . F (complex), F symmetric, ortho.
// One block processes IMGS_PER_BLK images; F staged in LDS once per block.
// 9 waves (288 thr), wave t -> 16x16 tile (t/3, t%3) of the padded 48x48 grid.
// EPI: 0=store complex f16, 1=CReLU+store, 2=store Re to f32 out.
// LDS tiles zero-padded to 48x64 -> no bounds checks in the WMMA loops.
// ---------------------------------------------------------------------------
template <int EPI>
__global__ __launch_bounds__(288)
void fft2_fused(const half_t* __restrict__ srcRe, const half_t* __restrict__ srcIm,
                const half_t* __restrict__ fre,   const half_t* __restrict__ fim,
                half_t* __restrict__ dstRe,       half_t* __restrict__ dstIm,
                float*  __restrict__ outF)
{
    __shared__ __align__(16) half_t lds[6 * TILE_ROWS * LDSW];
    half_t* ZIre = lds;
    half_t* ZIim = lds + 1 * TILE_ROWS * LDSW;
    half_t* ZMre = lds + 2 * TILE_ROWS * LDSW;  // Z*F, stored transposed [w][h]
    half_t* ZMim = lds + 3 * TILE_ROWS * LDSW;
    half_t* FMre = lds + 4 * TILE_ROWS * LDSW;
    half_t* FMim = lds + 5 * TILE_ROWS * LDSW;

    const int tid  = threadIdx.x;
    const int lane = tid & 31;
    const int wave = tid >> 5;            // 0..8
    const int mt = wave / 3, nt = wave % 3;
    const int l16 = lane & 15;
    const bool hi = lane >= 16;

    // zero all planes once (padding regions must stay 0), then stage F once
    for (int i = tid; i < 3 * TILE_ROWS * LDSW; i += 288) ((unsigned int*)lds)[i] = 0u;
    __syncthreads();
    for (int i = tid; i < FREQ / 2; i += 288) {
        int e2 = i * 2, r = e2 / IMG, c = e2 % IMG;
        *(unsigned int*)&FMre[r * LDSW + c] = ((const unsigned int*)fre)[i];
        *(unsigned int*)&FMim[r * LDSW + c] = ((const unsigned int*)fim)[i];
    }

    const int mA = mt * 16 + l16;
    const int nB = nt * 16 + l16;
    const int mBase = mt * 16 + (hi ? 8 : 0);

    for (int g = 0; g < IMGS_PER_BLK; ++g) {
        const int imgBase = (blockIdx.x * IMGS_PER_BLK + g) * FREQ;

        // stage this image (dword pairs never cross a row: 40 is even)
        for (int i = tid; i < FREQ / 2; i += 288) {
            int e2 = i * 2, r = e2 / IMG, c = e2 % IMG;
            *(unsigned int*)&ZIre[r * LDSW + c] = ((const unsigned int*)(srcRe + imgBase))[i];
            *(unsigned int*)&ZIim[r * LDSW + c] = ((const unsigned int*)(srcIm + imgBase))[i];
        }
        __syncthreads();

        // ---- stage 1: Z1[h,v] = sum_w Z[h,w] F[w,v]  (B[k=w][n=v] = FM[v][w], symm)
        {
            v8f accR = {}, accI = {};
#pragma unroll
            for (int ks = 0; ks < 64; ks += 32) {
                int ka = ks + (hi ? 8 : 0);
                int kb = ks + (hi ? 8 : 0);
                v16h ar = ld_fragA(ZIre + mA * LDSW, ka);
                v16h ai = ld_fragA(ZIim + mA * LDSW, ka);
                v16h br = ld_fragB(FMre + nB * LDSW, kb);
                v16h bi = ld_fragB(FMim + nB * LDSW, kb);
                v16h ain = -ai;
                CWMMA(ar, ai, ain, br, bi, accR, accI)
            }
            *(v8h*)&ZMre[nB * LDSW + mBase] = packv(accR);   // transposed store
            *(v8h*)&ZMim[nB * LDSW + mBase] = packv(accI);
        }
        __syncthreads();

        // ---- stage 2: Z2[u,w] = sum_h F[u,h] Z1[h,w]  (B[k=h][n=w] = ZM[w][h])
        {
            v8f accR = {}, accI = {};
#pragma unroll
            for (int ks = 0; ks < 64; ks += 32) {
                int ka = ks + (hi ? 8 : 0);
                int kb = ks + (hi ? 8 : 0);
                v16h ar = ld_fragA(FMre + mA * LDSW, ka);
                v16h ai = ld_fragA(FMim + mA * LDSW, ka);
                v16h br = ld_fragB(ZMre + nB * LDSW, kb);
                v16h bi = ld_fragB(ZMim + nB * LDSW, kb);
                v16h ain = -ai;
                CWMMA(ar, ai, ain, br, bi, accR, accI)
            }
#pragma unroll
            for (int r = 0; r < 8; ++r) {
                int m = mBase + r;
                if (m < IMG && nB < IMG) {
                    float re = accR[r], im = accI[r];
                    if (EPI == 1) { re = fmaxf(re, 0.f); im = fmaxf(im, 0.f); }
                    int off = imgBase + m * IMG + nB;
                    if (EPI == 2) outF[off] = re;
                    else { dstRe[off] = (half_t)re; dstIm[off] = (half_t)im; }
                }
            }
        }
        __syncthreads();   // protect ZM before next image's stage 1
    }
}

// ---------------------------------------------------------------------------
// Channel mix: per frequency bin f, C[b,o] = sum_i A[b,i] * W[i,o] (complex).
// Packed weights -> unconditional aligned b128 B loads; zero-padding makes
// all out-of-range K/N contributions exactly 0 (no bounds logic in the loop).
// 4 waves/block, one 16-row M tile per wave, A fragments reused over N tiles.
// ---------------------------------------------------------------------------
__global__ __launch_bounds__(128)
void mix_kernel(const half_t* __restrict__ Are, const half_t* __restrict__ Aim,
                const half_t* __restrict__ Wp,
                half_t* __restrict__ Cre, half_t* __restrict__ Cim,
                int ci, int co)
{
    const int lane = threadIdx.x & 31;
    const int wave = threadIdx.x >> 5;
    const int mt   = blockIdx.x * 4 + wave;       // 16 M tiles total
    const int f    = blockIdx.z;
    const int l16  = lane & 15;
    const bool hi  = lane >= 16;
    const int mA   = mt * 16 + l16;               // batch row, always < 256
    const int kA0  = hi ? 8 : 0;
    const int kB0  = hi ? 8 : 0;
    const int aRow = mA * ci * FREQ + f;

    const int nk = (ci > 32) ? 2 : 1;

    // A fragments (activations, strided u16 loads; address clamped, value
    // irrelevant when k >= ci because the matching Wp rows are zero)
    v16h arF[2], aiF[2], anF[2];
#pragma unroll
    for (int ks = 0; ks < 2; ++ks) {
        if (ks < nk) {
#pragma unroll
            for (int e = 0; e < 16; ++e) {
                int kk = ks * 32 + kA0 + (e < 8 ? e : e + 8);
                int kc = kk < ci ? kk : ci - 1;
                int off = aRow + kc * FREQ;
                arF[ks][e] = Are[off];
                aiF[ks][e] = Aim[off];
            }
            anF[ks] = -aiF[ks];
        }
    }

    const half_t* wr = Wp + f * WP_BIN;
    const half_t* wi = wr + WP_PLANE;
    const int nTiles = (co + 15) / 16;
    for (int nt = 0; nt < nTiles; ++nt) {
        const int nB = nt * 16 + l16;             // < 48, rows exist (padded)
        v8f accR = {}, accI = {};
#pragma unroll
        for (int ks = 0; ks < 2; ++ks) {
            if (ks < nk) {
                int kb = ks * 32 + kB0;
                v16h br = ld_fragB(wr + nB * LDSW, kb);
                v16h bi = ld_fragB(wi + nB * LDSW, kb);
                CWMMA(arF[ks], aiF[ks], anF[ks], br, bi, accR, accI)
            }
        }
        const int mBase = mt * 16 + (hi ? 8 : 0);
#pragma unroll
        for (int r = 0; r < 8; ++r) {
            int m = mBase + r;
            if (nB < co) {
                int off = (m * co + nB) * FREQ + f;
                Cre[off] = (half_t)accR[r];
                Cim[off] = (half_t)accI[r];
            }
        }
    }
}

// ---------------------------------------------------------------------------
extern "C" void kernel_launch(void* const* d_in, const int* in_sizes, int n_in,
                              void* d_out, int out_size, void* d_ws, size_t ws_size,
                              hipStream_t stream)
{
    const float* x = (const float*)d_in[0];
    const float* W[6];
    for (int i = 0; i < 6; ++i) W[i] = (const float*)d_in[1 + i];

    const int Ci[6] = {1, 40, 40, 40, 40, 40};
    const int Co[6] = {40, 40, 40, 40, 40, 1};

    const size_t PLANE = (size_t)BATCH * IMG * FREQ;   // 16,384,000 halves
    half_t* base   = (half_t*)d_ws;
    half_t* bufRe[2] = { base,             base + 2 * PLANE };
    half_t* bufIm[2] = { base + PLANE,     base + 3 * PLANE };
    half_t* fmat   = base + 4 * PLANE;                 // 3*FREQ halves
    half_t* wp     = base + 4 * PLANE + 8192;          // 16B-aligned, 9.83M halves

    init_dft_kernel<<<dim3((FREQ + 255) / 256), dim3(256), 0, stream>>>(fmat);
    pack_kernel<<<dim3((BATCH * FREQ + 255) / 256), dim3(256), 0, stream>>>(
        x, bufRe[0], bufIm[0], BATCH * FREQ);

    int s = 0;
    for (int l = 0; l < 6; ++l) {
        const int ci = Ci[l], co = Co[l];
        const int NI = BATCH * ci;
        const int NO = BATCH * co;

        // repack this layer's weights (stream order protects reuse of wp)
        repack_w_kernel<<<dim3((FREQ * TILE_ROWS + 255) / 256), dim3(256), 0, stream>>>(
            W[l], wp, ci, co);

        // fft2 (fused, 4 images per block)
        fft2_fused<0><<<dim3(NI / IMGS_PER_BLK), dim3(288), 0, stream>>>(
            bufRe[s], bufIm[s], fmat, fmat + FREQ,
            bufRe[1 - s], bufIm[1 - s], (float*)nullptr);
        s ^= 1;

        // channel mix per frequency bin: [256,ci] x [ci,co]
        mix_kernel<<<dim3(4, 1, FREQ), dim3(128), 0, stream>>>(
            bufRe[s], bufIm[s], wp, bufRe[1 - s], bufIm[1 - s], ci, co);
        s ^= 1;

        // ifft2 (fused, conj F) with fused epilogue
        if (l < 5) {
            fft2_fused<1><<<dim3(NO / IMGS_PER_BLK), dim3(288), 0, stream>>>(   // CReLU
                bufRe[s], bufIm[s], fmat, fmat + 2 * FREQ,
                bufRe[1 - s], bufIm[1 - s], (float*)nullptr);
        } else {
            fft2_fused<2><<<dim3(NO / IMGS_PER_BLK), dim3(288), 0, stream>>>(   // Re -> f32
                bufRe[s], bufIm[s], fmat, fmat + 2 * FREQ,
                bufRe[1 - s], bufIm[1 - s], (float*)d_out);
        }
        s ^= 1;
    }
}
